// MultiHeadAttentionBlock_2044404433004
// MI455X (gfx1250) — matrix operand order repaired and checked by reading, not compiled, not run on hardware
//
#include <hip/hip_runtime.h>
#include <hip/hip_bf16.h>

#define D_MODEL 1024
#define NHEAD   16
#define DK      64
#define SEQ     2048
#define BATCH   4

typedef __bf16 v16bf __attribute__((ext_vector_type(16)));
typedef __bf16 v8bf  __attribute__((ext_vector_type(8)));
typedef float  v8f   __attribute__((ext_vector_type(8)));
typedef float  v4f   __attribute__((ext_vector_type(4)));

// Convert 16 floats (p0[0..7], p1[0..7]) to a bf16 x16 fragment.
static __device__ inline v16bf bf16_from_f32x16(const float* p0, const float* p1) {
  v4f a0 = *(const v4f*)(p0);
  v4f a1 = *(const v4f*)(p0 + 4);
  v4f b0 = *(const v4f*)(p1);
  v4f b1 = *(const v4f*)(p1 + 4);
  v16bf r;
#pragma unroll
  for (int i = 0; i < 4; ++i) {
    r[i]      = (__bf16)a0[i];
    r[i + 4]  = (__bf16)a1[i];
    r[i + 8]  = (__bf16)b0[i];
    r[i + 12] = (__bf16)b1[i];
  }
  return r;
}

// Load 16 bf16 as two contiguous 8-element (16B) chunks.
static __device__ inline v16bf bf16_load16(const __bf16* p0, const __bf16* p1) {
  v8bf lo = *(const v8bf*)p0;
  v8bf hi = *(const v8bf*)p1;
  return __builtin_shufflevector(lo, hi, 0, 1, 2, 3, 4, 5, 6, 7, 8, 9, 10, 11, 12, 13, 14, 15);
}

static __device__ inline v8f wmma_bf16(v16bf a, v16bf b, v8f c) {
  return __builtin_amdgcn_wmma_f32_16x16x32_bf16(false, a, false, b, (short)0, c, false, false);
}

// C = A[M,K] * W[N,K]^T  (M=8192, N=K=1024).  One wave computes a 32x64 tile
// (2 A fragments x 4 B fragments = 8 WMMAs per K-step; each W fragment reused 2x).
// MODE 0: bf16 out, layout [B][H][S][DK]   (Q)
// MODE 1: bf16 out, layout [B][H][S][DK]   (K)
// MODE 2: bf16 out, layout [B][H][DK][S]   (V transposed)
// MODE 3: f32  out, layout [M][N]          (final projection)
// ABF: A is bf16 (true) or f32 (false)
template <int MODE, bool ABF>
__global__ __launch_bounds__(256) void gemm_wmma_kernel(const void* __restrict__ Av,
                                                        const float* __restrict__ W,
                                                        void* __restrict__ Outv) {
  const int lane = threadIdx.x & 31;
  const int wave = threadIdx.x >> 5;
  const int ln16 = lane & 15;
  const bool hi  = lane >= 16;
  const int rowBase = blockIdx.x * 256 + wave * 32;
  const int colBase = blockIdx.y * 64;

  const int m0    = rowBase + ln16;        // A-fragment 0 row
  const int m1    = rowBase + 16 + ln16;   // A-fragment 1 row
  const int kaoff = hi ? 8 : 0;            // A-fragment K packing (16-bit A layout)
  const int kboff = hi ? 8 : 0;            // B-fragment K packing (16-bit B layout)

  v8f acc[8];
#pragma unroll
  for (int c = 0; c < 8; ++c) acc[c] = {};

  for (int k0 = 0; k0 < D_MODEL; k0 += 32) {
    v16bf af0, af1;
    if (ABF) {
      const __bf16* a0 = (const __bf16*)Av + (size_t)m0 * D_MODEL + k0 + kaoff;
      const __bf16* a1 = (const __bf16*)Av + (size_t)m1 * D_MODEL + k0 + kaoff;
      af0 = bf16_load16(a0, a0 + 16);
      af1 = bf16_load16(a1, a1 + 16);
    } else {
      const float* a0 = (const float*)Av + (size_t)m0 * D_MODEL + k0 + kaoff;
      const float* a1 = (const float*)Av + (size_t)m1 * D_MODEL + k0 + kaoff;
      af0 = bf16_from_f32x16(a0, a0 + 16);
      af1 = bf16_from_f32x16(a1, a1 + 16);
    }
#pragma unroll
    for (int c = 0; c < 4; ++c) {
      const int n = colBase + 16 * c + ln16;  // B-fragment column = W row (contiguous K)
      const float* wr = W + (size_t)n * D_MODEL + k0 + kboff;
      v16bf bf = bf16_from_f32x16(wr, wr + 16);
      acc[c]     = wmma_bf16(af0, bf, acc[c]);
      acc[4 + c] = wmma_bf16(af1, bf, acc[4 + c]);
    }
  }

#pragma unroll
  for (int half = 0; half < 2; ++half) {
    const int rBase = rowBase + 16 * half;
#pragma unroll
    for (int c = 0; c < 4; ++c) {
      const int n = colBase + 16 * c + ln16;
#pragma unroll
      for (int r = 0; r < 8; ++r) {
        const int row = rBase + r + (hi ? 8 : 0);
        const float val = acc[4 * half + c][r];
        if (MODE == 3) {
          ((float*)Outv)[(size_t)row * D_MODEL + n] = val;
        } else {
          const int b = row >> 11, s = row & (SEQ - 1);
          const int h = n >> 6, d = n & (DK - 1);
          __bf16* o = (__bf16*)Outv;
          if (MODE == 2)
            o[((size_t)(b * NHEAD + h) * DK + d) * SEQ + s] = (__bf16)val;
          else
            o[((size_t)(b * NHEAD + h) * SEQ + s) * DK + d] = (__bf16)val;
        }
      }
    }
  }
}

// Flash-attention style: one wave owns 16 query rows of one (b,h).
// Streams 64 keys per iteration: 8 score WMMAs + online softmax + 8 PV WMMAs.
__global__ __launch_bounds__(256) void attn_wmma_kernel(const __bf16* __restrict__ Qb,
                                                        const __bf16* __restrict__ Kb,
                                                        const __bf16* __restrict__ VTb,
                                                        const int* __restrict__ mask,
                                                        __bf16* __restrict__ Ob) {
  __shared__ __bf16 Plds[8 * 16 * 64];   // per-wave 16x64 P tile (C-layout -> A-layout relay)
  const int lane = threadIdx.x & 31;
  const int wave = threadIdx.x >> 5;
  const int ln16 = lane & 15;
  const bool hi  = lane >= 16;
  __bf16* P = Plds + wave * (16 * 64);

  const int t  = blockIdx.x * 8 + wave;  // 0..8191
  const int qt = t & 127;
  const int h  = (t >> 7) & 15;
  const int b  = t >> 11;
  const int qbase = qt * 16;

  const size_t headOff = (size_t)(b * NHEAD + h) * SEQ * DK;
  const __bf16* Qh  = Qb + headOff;     // [S][DK]
  const __bf16* Kh  = Kb + headOff;     // [S][DK]
  const __bf16* VTh = VTb + headOff;    // [DK][S]
  const int* maskB  = mask + (size_t)b * SEQ * SEQ;

  const int kaoff = hi ? 8 : 0;
  const int kboff = hi ? 8 : 0;

  // Loop-invariant Q fragments (d = 0..31 and 32..63)
  const __bf16* qrow = Qh + (size_t)(qbase + ln16) * DK;
  v16bf qa0 = bf16_load16(qrow + kaoff, qrow + kaoff + 16);
  v16bf qa1 = bf16_load16(qrow + 32 + kaoff, qrow + 32 + kaoff + 16);

  float m_run[8], l_run[8];
  v8f o[4];
#pragma unroll
  for (int r = 0; r < 8; ++r) { m_run[r] = -1e30f; l_run[r] = 0.0f; }
#pragma unroll
  for (int c = 0; c < 4; ++c) o[c] = {};

  for (int j = 0; j < SEQ; j += 64) {
    // Score tiles: cs[f] covers keys j+16f .. j+16f+15
    v8f cs[4];
#pragma unroll
    for (int f = 0; f < 4; ++f) {
      const __bf16* kr = Kh + (size_t)(j + 16 * f + ln16) * DK + kboff;
      v16bf kbl = bf16_load16(kr, kr + 16);
      v16bf kbh = bf16_load16(kr + 32, kr + 48);
      v8f c = {};
      c = wmma_bf16(qa0, kbl, c);
      c = wmma_bf16(qa1, kbh, c);
      cs[f] = c;
    }

    // Scale + mask + row max (a score row lives across a 16-lane group at fixed VGPR index)
    float mx[8];
#pragma unroll
    for (int r = 0; r < 8; ++r) {
      const int q = qbase + r + (hi ? 8 : 0);
      const int* mrow = maskB + (size_t)q * SEQ + j + ln16;
      float tmx = -1e30f;
#pragma unroll
      for (int f = 0; f < 4; ++f) {
        const int mk = mrow[16 * f];
        cs[f][r] = mk ? cs[f][r] * 0.125f : -1e9f;
        tmx = fmaxf(tmx, cs[f][r]);
      }
#pragma unroll
      for (int x = 1; x < 16; x <<= 1) tmx = fmaxf(tmx, __shfl_xor(tmx, x, 32));
      mx[r] = tmx;
    }

    // Online softmax update
#pragma unroll
    for (int r = 0; r < 8; ++r) {
      const float mnew  = fmaxf(m_run[r], mx[r]);
      const float alpha = __expf(m_run[r] - mnew);
      m_run[r] = mnew;
      float rs = 0.0f;
#pragma unroll
      for (int f = 0; f < 4; ++f) {
        const float p = __expf(cs[f][r] - mnew);
        cs[f][r] = p;
        rs += p;
      }
#pragma unroll
      for (int x = 1; x < 16; x <<= 1) rs += __shfl_xor(rs, x, 32);
      l_run[r] = l_run[r] * alpha + rs;
#pragma unroll
      for (int c = 0; c < 4; ++c) o[c][r] *= alpha;
    }

    // Relay P (C-layout) -> A-layout via per-wave LDS tile (same-wave DS ops are in-order)
#pragma unroll
    for (int r = 0; r < 8; ++r) {
      const int row = r + (hi ? 8 : 0);
#pragma unroll
      for (int f = 0; f < 4; ++f)
        P[row * 64 + 16 * f + ln16] = (__bf16)cs[f][r];
    }
    __asm__ __volatile__("" ::: "memory");
    const __bf16* pr = P + ln16 * 64 + kaoff;
    v16bf pa0 = bf16_load16(pr, pr + 16);        // P keys j..j+31
    v16bf pa1 = bf16_load16(pr + 32, pr + 48);   // P keys j+32..j+63
    __asm__ __volatile__("" ::: "memory");

    // O += P (16x64) @ V (64x64); V columns contiguous thanks to VT layout
#pragma unroll
    for (int c = 0; c < 4; ++c) {
      const __bf16* vr = VTh + (size_t)(16 * c + ln16) * SEQ + j + kboff;
      v16bf vb0 = bf16_load16(vr, vr + 16);
      v16bf vb1 = bf16_load16(vr + 32, vr + 48);
      o[c] = wmma_bf16(pa0, vb0, o[c]);
      o[c] = wmma_bf16(pa1, vb1, o[c]);
    }
  }

  // Normalize and write bf16 attn-out in [B][S][D_MODEL] for the final GEMM
#pragma unroll
  for (int r = 0; r < 8; ++r) {
    const float inv = 1.0f / l_run[r];
    const int q = qbase + r + (hi ? 8 : 0);
#pragma unroll
    for (int c = 0; c < 4; ++c) {
      const int col = h * DK + 16 * c + ln16;
      Ob[(size_t)(b * SEQ + q) * D_MODEL + col] = (__bf16)(o[c][r] * inv);
    }
  }
}

extern "C" void kernel_launch(void* const* d_in, const int* in_sizes, int n_in,
                              void* d_out, int out_size, void* d_ws, size_t ws_size,
                              hipStream_t stream) {
  (void)in_sizes; (void)n_in; (void)out_size; (void)ws_size;
  const float* q   = (const float*)d_in[0];
  const float* k   = (const float*)d_in[1];
  const float* v   = (const float*)d_in[2];
  const int*  mask = (const int*)d_in[3];
  const float* w_q = (const float*)d_in[4];
  const float* w_k = (const float*)d_in[5];
  const float* w_v = (const float*)d_in[6];
  const float* w_o = (const float*)d_in[7];

  char* ws = (char*)d_ws;
  const size_t seg = (size_t)BATCH * NHEAD * SEQ * DK * sizeof(__bf16);  // 16 MiB
  __bf16* Qb  = (__bf16*)(ws);
  __bf16* Kb  = (__bf16*)(ws + seg);
  __bf16* VTb = (__bf16*)(ws + 2 * seg);
  __bf16* Ob  = (__bf16*)(ws + 3 * seg);

  dim3 blk(256);
  dim3 gGemm(32, 16);   // (8192/256) x (1024/64)

  gemm_wmma_kernel<0, false><<<gGemm, blk, 0, stream>>>((const void*)q, w_q, (void*)Qb);
  gemm_wmma_kernel<1, false><<<gGemm, blk, 0, stream>>>((const void*)k, w_k, (void*)Kb);
  gemm_wmma_kernel<2, false><<<gGemm, blk, 0, stream>>>((const void*)v, w_v, (void*)VTb);

  dim3 gAttn(1024);     // 8192 wave-tiles / 8 waves per block
  attn_wmma_kernel<<<gAttn, blk, 0, stream>>>(Qb, Kb, VTb, mask, Ob);

  gemm_wmma_kernel<3, true><<<gGemm, blk, 0, stream>>>((const void*)Ob, w_o, d_out);
}
